// VQPatchEncoder_74766790688840
// MI455X (gfx1250) — matrix operand order repaired and checked by reading, not compiled, not run on hardware
//
#include <hip/hip_runtime.h>
#include <hip/hip_bf16.h>

typedef __attribute__((ext_vector_type(16))) __bf16       v16bf;
typedef __attribute__((ext_vector_type(8)))  float        v8f;
typedef __attribute__((ext_vector_type(8)))  unsigned int v8u;

#define N_B   1024
#define N_P   64
#define N_D   192
#define N_K   4096
#define N_E   2048
#define ROWS  (N_B * N_P)          // 65536 (batch,patch) rows
#define NTILES (N_K / 16)          // 256 column tiles

// output layout (tuple flattened): z_real | z_vsa | indices | z_local
#define OFF_ZREAL  0
#define OFF_ZVSA   (N_B * N_E)                 // 2097152
#define OFF_IDX    (2 * N_B * N_E)             // 4194304
#define OFF_ZLOCAL (2 * N_B * N_E + ROWS)      // 4259840

__device__ __forceinline__ unsigned short f2bf(float f) {
    unsigned int u = __float_as_uint(f);
    return (unsigned short)((u + 0x7FFFu + ((u >> 16) & 1u)) >> 16);  // RNE
}

// B fragment: two contiguous 8-elt (16B) chunks per lane, K and K+16
__device__ __forceinline__ v16bf loadB16(const unsigned short* p) {
    uint4 lo = *(const uint4*)p;
    uint4 hi = *(const uint4*)(p + 16);
    v8u r;
    r[0] = lo.x; r[1] = lo.y; r[2] = lo.z; r[3] = lo.w;
    r[4] = hi.x; r[5] = hi.y; r[6] = hi.z; r[7] = hi.w;
    return __builtin_bit_cast(v16bf, r);
}

// A fragment: two contiguous 8-elt (16B) chunks per lane, K and K+16
__device__ __forceinline__ v16bf loadA16(const unsigned short* p) {
    uint4 lo = *(const uint4*)p;         // K .. K+7
    uint4 hi = *(const uint4*)(p + 16);  // K+16 .. K+23
    v8u r;
    r[0] = lo.x; r[1] = lo.y; r[2] = lo.z; r[3] = lo.w;
    r[4] = hi.x; r[5] = hi.y; r[6] = hi.z; r[7] = hi.w;
    return __builtin_bit_cast(v16bf, r);
}

// ---- Kernel 1: patchify + L2-normalize + convert to bf16 (1 wave / patch) ----
__global__ void __launch_bounds__(256)
k_patchify(const float* __restrict__ pixels, unsigned short* __restrict__ out) {
    int wave = threadIdx.x >> 5;
    int lane = threadIdx.x & 31;
    int gp   = blockIdx.x * 8 + wave;          // global patch row < 65536
    int b    = gp >> 6;
    int p    = gp & 63;
    int pr   = p >> 3, pc = p & 7;
    const float* base = pixels + (size_t)b * 3 * 64 * 64;

    float v[6];
    float ss = 0.f;
#pragma unroll
    for (int i = 0; i < 6; ++i) {
        int d   = lane + 32 * i;               // d < 192
        int c   = d >> 6;                      // channel
        int rem = d & 63;
        int py  = rem >> 3, px = rem & 7;
        float x = base[((size_t)c * 64 + (pr * 8 + py)) * 64 + (pc * 8 + px)];
        v[i] = x;
        ss += x * x;
    }
#pragma unroll
    for (int m = 16; m >= 1; m >>= 1) ss += __shfl_xor(ss, m, 32);
    float scale = 1.0f / fmaxf(sqrtf(ss), 1e-8f);

    unsigned short* o = out + (size_t)gp * N_D;
#pragma unroll
    for (int i = 0; i < 6; ++i) o[lane + 32 * i] = f2bf(v[i] * scale);
}

// ---- Kernel 1b: codebook f32 -> bf16 (4 elems / thread) ----
__global__ void __launch_bounds__(256)
k_cvt_codebook(const float* __restrict__ cb, unsigned short* __restrict__ out) {
    int t = blockIdx.x * blockDim.x + threadIdx.x;     // < 196608
    float4 f = ((const float4*)cb)[t];
    unsigned int lo = (unsigned int)f2bf(f.x) | ((unsigned int)f2bf(f.y) << 16);
    unsigned int hi = (unsigned int)f2bf(f.z) | ((unsigned int)f2bf(f.w) << 16);
    ((uint2*)out)[t] = make_uint2(lo, hi);
}

// ---- Kernel 2: bf16 WMMA GEMM (sim) with fused argmax, software-pipelined ----
// 8 waves / block, each wave owns one 16-row M-tile and sweeps 256 N-tiles.
// B tiles are double-buffered in registers: loads for tile nt+1 stay in
// flight while the 6 WMMAs of tile nt execute (partial s_wait_loadcnt).
__global__ void __launch_bounds__(256)
k_sim_argmax(const unsigned short* __restrict__ A,   // patches bf16 [65536,192]
             const unsigned short* __restrict__ Bm,  // codebook bf16 [4096,192]
             int* __restrict__ idx_ws,
             float* __restrict__ idx_out) {
    int wave    = threadIdx.x >> 5;
    int lane    = threadIdx.x & 31;
    int mTile   = blockIdx.x * 8 + wave;       // < 4096
    int m0      = mTile * 16;
    int colLane = lane & 15;
    int half    = lane >> 4;

    // A fragments for K = 0..191 (6 x 16x16x32 steps), held for all N-tiles
    v16bf a[6];
#pragma unroll
    for (int kb = 0; kb < 6; ++kb) {
        int rowA = m0 + colLane;                       // M = lane&15 in both halves
        int koff = kb * 32 + half * 8;                 // ISA A-layout half-lane K offset
        a[kb] = loadA16(A + (size_t)rowA * N_D + koff);
    }

    float best[8];
    int   bidx[8];
#pragma unroll
    for (int r = 0; r < 8; ++r) { best[r] = -__builtin_inff(); bidx[r] = 0; }

    // per-lane base pointer into B: column = colLane of tile 0, half-lane K offset
    const unsigned short* bp0 = Bm + (size_t)colLane * N_D + half * 8;
    const size_t tileStride = (size_t)16 * N_D;        // 16 codes per N-tile

    v16bf b0[6], b1[6];
#pragma unroll
    for (int kb = 0; kb < 6; ++kb) b0[kb] = loadB16(bp0 + kb * 32);   // tile 0

    for (int nt = 0; nt < NTILES; nt += 2) {
        // issue loads for tile nt+1 into b1 (overlaps with compute of nt)
        {
            const unsigned short* bp = bp0 + (size_t)(nt + 1) * tileStride;
#pragma unroll
            for (int kb = 0; kb < 6; ++kb) b1[kb] = loadB16(bp + kb * 32);
        }
        // compute tile nt from b0
        {
            v8f c = {};
#pragma unroll
            for (int kb = 0; kb < 6; ++kb)
                c = __builtin_amdgcn_wmma_f32_16x16x32_bf16(
                        false, a[kb], false, b0[kb], (short)0, c, false, false);
            int colB = nt * 16 + colLane;
#pragma unroll
            for (int r = 0; r < 8; ++r)
                if (c[r] > best[r]) { best[r] = c[r]; bidx[r] = colB; }
        }
        // issue loads for tile nt+2 into b0 (clamped tail prefetch, result unused)
        {
            int ntp = (nt + 2 < NTILES) ? (nt + 2) : (NTILES - 1);
            const unsigned short* bp = bp0 + (size_t)ntp * tileStride;
#pragma unroll
            for (int kb = 0; kb < 6; ++kb) b0[kb] = loadB16(bp + kb * 32);
        }
        // compute tile nt+1 from b1
        {
            v8f c = {};
#pragma unroll
            for (int kb = 0; kb < 6; ++kb)
                c = __builtin_amdgcn_wmma_f32_16x16x32_bf16(
                        false, a[kb], false, b1[kb], (short)0, c, false, false);
            int colB = (nt + 1) * 16 + colLane;
#pragma unroll
            for (int r = 0; r < 8; ++r)
                if (c[r] > best[r]) { best[r] = c[r]; bidx[r] = colB; }
        }
    }

    // reduce (max, first-index) across the 16 lanes sharing each row
#pragma unroll
    for (int r = 0; r < 8; ++r) {
        float bv = best[r]; int bi = bidx[r];
#pragma unroll
        for (int m = 8; m >= 1; m >>= 1) {
            float ov = __shfl_xor(bv, m, 16);
            int   oi = __shfl_xor(bi, m, 16);
            if (ov > bv || (ov == bv && oi < bi)) { bv = ov; bi = oi; }
        }
        best[r] = bv; bidx[r] = bi;
    }
    if (colLane == 0) {                                // lanes 0 and 16
#pragma unroll
        for (int r = 0; r < 8; ++r) {
            int row = m0 + half * 8 + r;               // VGPR r holds row r + 8*half
            idx_ws[row]  = bidx[r];
            idx_out[row] = (float)bidx[r];
        }
    }
}

// ---- Kernel 3: embedding gather means + binary VSA (1 block / batch) ----
__global__ void __launch_bounds__(256)
k_gather(const float* __restrict__ emb,
         const float* __restrict__ vsa,
         const float* __restrict__ roles,
         const int* __restrict__ idx,
         float* __restrict__ out) {
    __shared__ int s_idx[N_P];
    int b = blockIdx.x;
    int t = threadIdx.x;
    if (t < N_P) s_idx[t] = idx[b * N_P + t];
    __syncthreads();

    int base = t * 8;                                  // 256 threads * 8 dims = 2048
    float accR[8] = {}; float accL[8] = {};
    for (int p = 0; p < N_P; ++p) {
        int k = s_idx[p];
        const float4* e = (const float4*)(emb + (size_t)k * N_E + base);
        float ev[8];
        *(float4*)&ev[0] = e[0];
        *(float4*)&ev[4] = e[1];
        int pr = p >> 3, pc = p & 7;
        bool ag = (pr >= 3) & (pr <= 5) & (pc >= 3) & (pc <= 5);   // AGENT 3x3
#pragma unroll
        for (int j = 0; j < 8; ++j) {
            accR[j] += ev[j];
            if (ag) accL[j] += ev[j];
        }
    }

    // VSA over CENTRAL 4x4: bound = (cb_vsa XOR role), majority over 16
    int cnt[8] = {};
#pragma unroll
    for (int rr = 2; rr < 6; ++rr) {
#pragma unroll
        for (int cc = 2; cc < 6; ++cc) {
            int p = rr * 8 + cc;
            int k = s_idx[p];
            const float4* v = (const float4*)(vsa   + (size_t)k * N_E + base);
            const float4* q = (const float4*)(roles + (size_t)p * N_E + base);
            float vv[8], qq[8];
            *(float4*)&vv[0] = v[0]; *(float4*)&vv[4] = v[1];
            *(float4*)&qq[0] = q[0]; *(float4*)&qq[4] = q[1];
#pragma unroll
            for (int j = 0; j < 8; ++j)
                cnt[j] += ((int)vv[j]) ^ ((int)qq[j]);
        }
    }

    float* zr = out + OFF_ZREAL  + (size_t)b * N_E + base;
    float* zv = out + OFF_ZVSA   + (size_t)b * N_E + base;
    float* zl = out + OFF_ZLOCAL + (size_t)b * N_E + base;
#pragma unroll
    for (int j = 0; j < 8; ++j) {
        zr[j] = accR[j] * (1.0f / 64.0f);
        zl[j] = accL[j] * (1.0f / 9.0f);
        zv[j] = (cnt[j] > 8) ? 1.0f : 0.0f;
    }
}

extern "C" void kernel_launch(void* const* d_in, const int* in_sizes, int n_in,
                              void* d_out, int out_size, void* d_ws, size_t ws_size,
                              hipStream_t stream) {
    const float* pixels    = (const float*)d_in[0];
    const float* codebook  = (const float*)d_in[1];
    const float* embed     = (const float*)d_in[2];
    const float* cb_vsa    = (const float*)d_in[3];
    const float* roles     = (const float*)d_in[4];
    float* out = (float*)d_out;

    char* ws = (char*)d_ws;
    unsigned short* patches_bf = (unsigned short*)ws;                              // 24 MB
    unsigned short* cb_bf      = (unsigned short*)(ws + (size_t)ROWS * N_D * 2);   // 1.5 MB
    int*            idx_ws     = (int*)(ws + (size_t)ROWS * N_D * 2
                                           + (size_t)N_K  * N_D * 2);              // 256 KB

    k_patchify    <<<ROWS / 8,               256, 0, stream>>>(pixels, patches_bf);
    k_cvt_codebook<<<(N_K * N_D / 4) / 256,  256, 0, stream>>>(codebook, cb_bf);
    k_sim_argmax  <<<ROWS / 16 / 8,          256, 0, stream>>>(patches_bf, cb_bf,
                                                               idx_ws, out + OFF_IDX);
    k_gather      <<<N_B,                    256, 0, stream>>>(embed, cb_vsa, roles,
                                                               idx_ws, out);
}
